// SSMLayer_53549652247054
// MI455X (gfx1250) — compile-verified, matrix operand order checked by reading, not run on hardware
//
#include <hip/hip_runtime.h>
#include <hip/hip_bf16.h>

#define D_MODEL 1024
#define BATCH   4
#define T_LEN   8192
#define LN_EPS  1e-5f

typedef __attribute__((ext_vector_type(16))) __bf16   v16bf;
typedef __attribute__((ext_vector_type(8)))  __bf16   v8bf;
typedef __attribute__((ext_vector_type(4)))  __bf16   v4bf;
typedef __attribute__((ext_vector_type(8)))  float    v8f;
typedef __attribute__((ext_vector_type(4)))  float    v4f;
typedef __attribute__((ext_vector_type(4)))  unsigned v4u;
typedef __attribute__((ext_vector_type(8)))  int      v8i;
typedef __attribute__((ext_vector_type(4)))  int      v4i;

#if defined(__has_builtin)
#  if __has_builtin(__builtin_amdgcn_tensor_load_to_lds)
#    define HAVE_TDM 1
#  endif
#endif
#ifndef HAVE_TDM
#  define HAVE_TDM 0
#endif

// ---------------- f32 -> bf16 conversion (x4 vectorized) ----------------
__global__ __launch_bounds__(256)
void cvt_f32_bf16(const float* __restrict__ in, __bf16* __restrict__ out, long n)
{
    long i = ((long)blockIdx.x * 256 + threadIdx.x) * 4;
    if (i >= n) return;
    v4f x = *reinterpret_cast<const v4f*>(in + i);
    v4bf y;
    y[0] = (__bf16)x[0]; y[1] = (__bf16)x[1];
    y[2] = (__bf16)x[2]; y[3] = (__bf16)x[3];
    *reinterpret_cast<v4bf*>(out + i) = y;
}

// ---------------- TDM: DMA a rows x 32 bf16 tile (row stride = ld) to LDS ----------
#if HAVE_TDM
__device__ __forceinline__
void tdm_load_2d(unsigned lds_addr, const void* gsrc,
                 unsigned tile_rows, unsigned tensor_rows, unsigned ld)
{
    const unsigned long long ga = (unsigned long long)(uintptr_t)gsrc;
    // D# group 0: count=1 | lds_addr | global_addr | type=2 ("image")
    v4u g0;
    g0.x = 1u;
    g0.y = lds_addr;
    g0.z = (unsigned)ga;
    g0.w = (unsigned)((ga >> 32) & 0x01FFFFFFu) | (2u << 30);
    // D# group 1: data_size=1(2B); tensor_dim0=ld; tensor_dim1=tensor_rows;
    //             tile_dim0=32; tile_dim1=tile_rows; tensor_dim0_stride=ld
    v8i g1;
    g1[0] = (int)(1u << 16);
    g1[1] = (int)((ld & 0xFFFFu) << 16);
    g1[2] = (int)((ld >> 16) | ((tensor_rows & 0xFFFFu) << 16));
    g1[3] = (int)((tensor_rows >> 16) | (32u << 16));
    g1[4] = (int)(tile_rows & 0xFFFFu);
    g1[5] = (int)ld;
    g1[6] = 0;
    g1[7] = 0;
    const v4i z4 = (v4i)0;
#if defined(__clang_major__) && (__clang_major__ >= 23)
    __builtin_amdgcn_tensor_load_to_lds(g0, g1, z4, z4, (v8i)0, 0);
#else
    __builtin_amdgcn_tensor_load_to_lds(g0, g1, z4, z4, 0);
#endif
}
#endif

__device__ __forceinline__
void stage_tile(__bf16* lds_dst, const __bf16* gsrc,
                int tile_rows, int tensor_rows, int ld, int tid)
{
#if HAVE_TDM
    if (tid < 32)   // wave 0 issues the DMA; TENSORcnt is per-wave
        tdm_load_2d((unsigned)(uintptr_t)lds_dst, gsrc,
                    (unsigned)tile_rows, (unsigned)tensor_rows, (unsigned)ld);
#else
    for (int i = tid; i < tile_rows * 4; i += 256) {
        const int r = i >> 2, c = (i & 3) << 3;
        *reinterpret_cast<v8bf*>(lds_dst + r * 32 + c) =
            *reinterpret_cast<const v8bf*>(gsrc + (long)r * ld + c);
    }
#endif
}

__device__ __forceinline__ void wait_stage()
{
#if HAVE_TDM
    __builtin_amdgcn_s_wait_tensorcnt(0);
#endif
}

// ---------------- BF16 WMMA GEMM:  C[M][N] = A[M][K] * W[N][K]^T ----------------
// ISA 16-bit A-fragment layout (16x32, wave32):
//   lane l: row = l%16, half = l/16; elems 0..7 -> k = half*8+e, 8..15 -> +16.
// B fragment (32x16, out = A*W^T) reads ROWS of W with the same pattern.
union Frag {
    v16bf v;
    struct { v8bf lo, hi; } h;
};

__device__ __forceinline__ v16bf load_frag(const __bf16* __restrict__ rowp, int kb)
{
    Frag f;
    f.h.lo = *reinterpret_cast<const v8bf*>(rowp + kb);
    f.h.hi = *reinterpret_cast<const v8bf*>(rowp + kb + 16);
    return f.v;
}

__global__ __launch_bounds__(256)
void gemm_bf16_nt(const __bf16* __restrict__ A,   // [M][K]
                  const __bf16* __restrict__ W,   // [N][K]
                  float* __restrict__ C,          // [M][N]
                  int M, int N, int K)
{
    // double-buffered LDS tiles, filled by the Tensor Data Mover
    __shared__ __align__(16) __bf16 Asm[2][128 * 32];   // 16 KB
    __shared__ __align__(16) __bf16 Bsm[2][256 * 32];   // 32 KB

    const int tid  = threadIdx.x;
    const int lane = tid & 31;
    const int wave = tid >> 5;
    const int wm   = wave & 1;        // 2 waves along M
    const int wn   = wave >> 1;       // 4 waves along N
    const int half = lane >> 4;
    const int l16  = lane & 15;

    const int mblk = blockIdx.y * 128;
    const int nblk = blockIdx.x * 256;
    const int m0   = mblk + wm * 64;
    const int n0   = nblk + wn * 64;

    const __bf16* Ag = A + (long)mblk * K;
    const __bf16* Wg = W + (long)nblk * K;

    v8f acc[4][4] = {};

    // prologue: stage chunk 0 into buffer 0
    stage_tile(&Asm[0][0], Ag, 128, M, K, tid);
    stage_tile(&Bsm[0][0], Wg, 256, N, K, tid);
    wait_stage();
    __syncthreads();

    int cur = 0;
    for (int kc = 0; kc < K; kc += 32) {
        // stage next chunk while computing on the current one
        if (kc + 32 < K) {
            stage_tile(&Asm[cur ^ 1][0], Ag + kc + 32, 128, M, K, tid);
            stage_tile(&Bsm[cur ^ 1][0], Wg + kc + 32, 256, N, K, tid);
        }

        const __bf16* Ab = &Asm[cur][0];
        const __bf16* Bb = &Bsm[cur][0];
        const int kb = half * 8;
        v16bf a[4], b[4];
#pragma unroll
        for (int t = 0; t < 4; ++t)
            a[t] = load_frag(Ab + (wm * 64 + t * 16 + l16) * 32, kb);
#pragma unroll
        for (int t = 0; t < 4; ++t)
            b[t] = load_frag(Bb + (wn * 64 + t * 16 + l16) * 32, kb);
#pragma unroll
        for (int mt = 0; mt < 4; ++mt)
#pragma unroll
            for (int nt = 0; nt < 4; ++nt)
                acc[mt][nt] = __builtin_amdgcn_wmma_f32_16x16x32_bf16(
                    false, a[mt], false, b[nt],
                    (short)0, acc[mt][nt], false, false);

        wait_stage();        // TDM of next chunk complete (no-op on last iter)
        __syncthreads();     // publish LDS to all waves / retire ds reads
        cur ^= 1;
    }

    // C/D layout: VGPR v, lanes 0-15 -> M = v, lanes 16-31 -> M = v+8; N = lane%16
#pragma unroll
    for (int mt = 0; mt < 4; ++mt) {
#pragma unroll
        for (int nt = 0; nt < 4; ++nt) {
            const int mbase = m0 + mt * 16 + half * 8;
            const int n     = n0 + nt * 16 + l16;
#pragma unroll
            for (int v = 0; v < 8; ++v)
                C[(long)(mbase + v) * N + n] = acc[mt][nt][v];
        }
    }
}

// ---------------- diagonal SSM scan: s_t = a*s_{t-1} + ub_t ----------------
__global__ __launch_bounds__(256)
void ssm_scan(const float* __restrict__ ub,        // [B][T][D] f32
              const float* __restrict__ state0,    // [B][D]
              const float* __restrict__ log_a,     // [D]
              __bf16* __restrict__ states_bf,      // [B][T][D]
              float* __restrict__ final_state)     // [B][D]
{
    const int idx = blockIdx.x * 256 + threadIdx.x;    // 0 .. B*D-1
    const int b = idx >> 10;
    const int d = idx & (D_MODEL - 1);
    const float a = expf(log_a[d]);
    float s = state0[idx];
    const float* p = ub        + ((long)b * T_LEN) * D_MODEL + d;
    __bf16*      q = states_bf + ((long)b * T_LEN) * D_MODEL + d;
#pragma unroll 8
    for (int t = 0; t < T_LEN; ++t) {
        s = fmaf(a, s, p[(long)t * D_MODEL]);
        q[(long)t * D_MODEL] = (__bf16)s;
    }
    final_state[idx] = s;
}

// ---------------- in-place row LayerNorm over D=1024 ----------------
__global__ __launch_bounds__(256)
void layernorm_inplace(float* __restrict__ y,
                       const float* __restrict__ gamma,
                       const float* __restrict__ beta)
{
    __shared__ float rs[8], rq[8];
    float* p = y + (long)blockIdx.x * D_MODEL;
    const int t = threadIdx.x;
    float x[4], sum = 0.f, sq = 0.f;
#pragma unroll
    for (int i = 0; i < 4; ++i) {
        x[i] = p[t + i * 256];
        sum += x[i];
        sq  += x[i] * x[i];
    }
#pragma unroll
    for (int off = 16; off; off >>= 1) {
        sum += __shfl_xor(sum, off, 32);
        sq  += __shfl_xor(sq,  off, 32);
    }
    if ((t & 31) == 0) { rs[t >> 5] = sum; rq[t >> 5] = sq; }
    __syncthreads();
    sum = 0.f; sq = 0.f;
#pragma unroll
    for (int w = 0; w < 8; ++w) { sum += rs[w]; sq += rq[w]; }
    const float mu  = sum * (1.f / D_MODEL);
    const float var = sq  * (1.f / D_MODEL) - mu * mu;
    const float inv = rsqrtf(var + LN_EPS);
#pragma unroll
    for (int i = 0; i < 4; ++i) {
        const int d = t + i * 256;
        p[d] = (x[i] - mu) * inv * gamma[d] + beta[d];
    }
}

// ---------------- driver ----------------
extern "C" void kernel_launch(void* const* d_in, const int* in_sizes, int n_in,
                              void* d_out, int out_size, void* d_ws, size_t ws_size,
                              hipStream_t stream)
{
    const float* u      = (const float*)d_in[0];   // (B,T,D)
    const float* state0 = (const float*)d_in[1];   // (B,D)
    const float* log_a  = (const float*)d_in[2];   // (D,)
    const float* Wb     = (const float*)d_in[3];   // (D,D)
    const float* Wc     = (const float*)d_in[4];   // (D,D)
    const float* gamma  = (const float*)d_in[5];   // (D,)
    const float* beta   = (const float*)d_in[6];   // (D,)
    float* out = (float*)d_out;                    // [B*T*D ln_out][B*D final_state]

    const long MT  = (long)BATCH * T_LEN;          // 32768 rows
    const long NU  = MT * D_MODEL;                 // 33,554,432
    const long NW  = (long)D_MODEL * D_MODEL;      // 1,048,576

    // workspace layout (~196 MB)
    char* ws = (char*)d_ws;
    __bf16* u_bf  = (__bf16*)ws;                                 // 64 MB (reused as states_bf)
    __bf16* wb_bf = (__bf16*)(ws + NU * sizeof(__bf16));
    __bf16* wc_bf = (__bf16*)(ws + NU * sizeof(__bf16) + NW * sizeof(__bf16));
    float*  ub_f  = (float*) (ws + NU * sizeof(__bf16) + 2 * NW * sizeof(__bf16)); // 128 MB

    // 1) f32 -> bf16 staging
    cvt_f32_bf16<<<(int)(NU / 4 / 256), 256, 0, stream>>>(u,  u_bf,  NU);
    cvt_f32_bf16<<<(int)(NW / 4 / 256), 256, 0, stream>>>(Wb, wb_bf, NW);
    cvt_f32_bf16<<<(int)(NW / 4 / 256), 256, 0, stream>>>(Wc, wc_bf, NW);

    // 2) u_b = u @ Wb^T   (TDM-staged bf16 WMMA, f32 accumulate)
    dim3 ggrid(D_MODEL / 256, (unsigned)(MT / 128));
    gemm_bf16_nt<<<ggrid, 256, 0, stream>>>(u_bf, wb_bf, ub_f,
                                            (int)MT, D_MODEL, D_MODEL);

    // 3) recurrent scan; states stored bf16 into u_bf, final state f32 to tail
    ssm_scan<<<(BATCH * D_MODEL) / 256, 256, 0, stream>>>(
        ub_f, state0, log_a, u_bf, out + NU);

    // 4) y = states @ Wc^T -> directly into d_out
    gemm_bf16_nt<<<ggrid, 256, 0, stream>>>(u_bf, wc_bf, out,
                                            (int)MT, D_MODEL, D_MODEL);

    // 5) in-place LayerNorm over each row of y
    layernorm_inplace<<<(unsigned)MT, 256, 0, stream>>>(out, gamma, beta);
}
